// ExplicitRelationEncoder_17420387352925
// MI455X (gfx1250) — matrix operand order repaired and checked by reading, not compiled, not run on hardware
//
#include <hip/hip_runtime.h>
#include <hip/hip_bf16.h>
#include <cstdint>

// ---------------------------------------------------------------------------
// B=256, R=100, V_DIM=2048, Q_DIM=1024, OUT_DIM=F=1024, HEADS=16, dg=64,
// NONGT n=20, LABEL_NUM=11, DIRS=2, STEPS=1.
// ---------------------------------------------------------------------------

typedef __bf16 bf16_t;
typedef __bf16 v16bf __attribute__((ext_vector_type(16)));
typedef __bf16 v8bf  __attribute__((ext_vector_type(8)));
typedef float  v8f   __attribute__((ext_vector_type(8)));

union Frag {
  v16bf v;
  v8bf  h[2];
};

#define BM 128
#define BN 128
#define BK 32
#define LPITCH 40   // bf16 elems per LDS row (80B: 16B-aligned, stride-20 banking)

// ---------------------------------------------------------------------------
// bf16-WMMA GEMM:  C[M,N](f32) (+)= A[M,K](f32 -> bf16) @ Bt[N,K](bf16, transposed)
//   amap: 0 = identity rows, 1 = xk gather: row -> (row/20)*100 + row%20
//   mode: 0 = store f32, 1 = relu f32, 2 = accumulate f32, 3 = store bf16
// 256 threads = 8 waves; wave grid 4(M)x2(N); each wave: 2x4 16x16 WMMA accs.
// A fragment = 2x contiguous-8 K-runs; B fragment = same 2x contiguous-8 K-runs per
// column (Bs staged K-major) -> all LDS traffic is ds_load/ds_store_b128.
// ---------------------------------------------------------------------------
__global__ __launch_bounds__(256) void gemm_bf16_kernel(
    const float* __restrict__ A, int lda, int amap,
    const bf16_t* __restrict__ Bt, int ldbt, const float* __restrict__ bias,
    float* __restrict__ Cf, bf16_t* __restrict__ Cb,
    int M, int N, int K, int mode)
{
  __shared__ __align__(16) bf16_t As[BM * LPITCH];
  __shared__ __align__(16) bf16_t Bs[BN * LPITCH];

  const int tid   = threadIdx.x;
  const int lane  = tid & 31;
  const int wid   = tid >> 5;
  const int waveM = wid & 3;   // 4 waves along M (32 rows each)
  const int waveN = wid >> 2;  // 2 waves along N (64 cols each)
  const int blockM = blockIdx.x * BM;
  const int blockN = blockIdx.y * BN;

  v8f acc[2][4];
#pragma unroll
  for (int i = 0; i < 2; ++i)
#pragma unroll
    for (int j = 0; j < 4; ++j)
#pragma unroll
      for (int e = 0; e < 8; ++e) acc[i][j][e] = 0.0f;

  // A staging map: thread -> (row 0..127, K-half 0/16), 16 f32 -> 16 bf16
  const int arl = tid >> 1;
  const int acb = (tid & 1) << 4;
  int agr = blockM + arl;
  if (amap) agr = (agr / 20) * 100 + (agr % 20);   // xk row gather
  const float* aptr = A + (size_t)agr * lda + acb;

  // B staging map: thread -> (col 0..127, K-half 0/16), 16 bf16 contiguous
  const int bcl = tid >> 1;
  const int bkh = (tid & 1) << 4;
  const bf16_t* bptr = Bt + (size_t)(blockN + bcl) * ldbt + bkh;

  for (int k0 = 0; k0 < K; k0 += BK) {
    __syncthreads();
    // ---- stage A (f32 -> bf16), two b128 stores ----
    {
      const float4* p = (const float4*)(aptr + k0);
      float4 f0 = p[0], f1 = p[1], f2 = p[2], f3 = p[3];
      v8bf t0, t1;
      t0[0]=(bf16_t)f0.x; t0[1]=(bf16_t)f0.y; t0[2]=(bf16_t)f0.z; t0[3]=(bf16_t)f0.w;
      t0[4]=(bf16_t)f1.x; t0[5]=(bf16_t)f1.y; t0[6]=(bf16_t)f1.z; t0[7]=(bf16_t)f1.w;
      t1[0]=(bf16_t)f2.x; t1[1]=(bf16_t)f2.y; t1[2]=(bf16_t)f2.z; t1[3]=(bf16_t)f2.w;
      t1[4]=(bf16_t)f3.x; t1[5]=(bf16_t)f3.y; t1[6]=(bf16_t)f3.z; t1[7]=(bf16_t)f3.w;
      *(v8bf*)&As[arl * LPITCH + acb]     = t0;
      *(v8bf*)&As[arl * LPITCH + acb + 8] = t1;
    }
    // ---- stage B (bf16, K-major per column), two b128 stores ----
    {
      const uint4* p = (const uint4*)(bptr + k0);
      uint4 u0 = p[0], u1 = p[1];
      uint4* d = (uint4*)&Bs[bcl * LPITCH + bkh];
      d[0] = u0; d[1] = u1;
    }
    if (k0 + BK < K) {  // global_prefetch_b8 next tiles
      __builtin_prefetch(aptr + (k0 + BK), 0, 1);
      __builtin_prefetch(bptr + (k0 + BK), 0, 1);
    }
    __syncthreads();

    // ---- A fragments: two contiguous 8-elem K-runs per lane (2x b128) ----
    Frag af[2];
#pragma unroll
    for (int mt = 0; mt < 2; ++mt) {
      const int row = waveM * 32 + mt * 16 + (lane & 15);
      const int kb  = (lane < 16) ? 0 : 8;
      af[mt].h[0] = *(const v8bf*)&As[row * LPITCH + kb];
      af[mt].h[1] = *(const v8bf*)&As[row * LPITCH + 16 + kb];
    }
    // ---- B fragments: two contiguous 8-elem K-runs per lane (2x b128) ----
    Frag bf[4];
#pragma unroll
    for (int nt = 0; nt < 4; ++nt) {
      const int col = waveN * 64 + nt * 16 + (lane & 15);
      const int kb  = (lane < 16) ? 0 : 8;
      bf[nt].h[0] = *(const v8bf*)&Bs[col * LPITCH + kb];
      bf[nt].h[1] = *(const v8bf*)&Bs[col * LPITCH + 16 + kb];
    }
    // ---- 8 WMMAs ----
#pragma unroll
    for (int mt = 0; mt < 2; ++mt)
#pragma unroll
      for (int nt = 0; nt < 4; ++nt)
        acc[mt][nt] = __builtin_amdgcn_wmma_f32_16x16x32_bf16(
            false, af[mt].v, false, bf[nt].v, (short)0, acc[mt][nt], false, false);
  }

  // ---- epilogue (C layout: VGPR i -> M = i + (lane<16 ? 0 : 8), N = lane&15) ----
#pragma unroll
  for (int mt = 0; mt < 2; ++mt) {
#pragma unroll
    for (int nt = 0; nt < 4; ++nt) {
      const int col = blockN + waveN * 64 + nt * 16 + (lane & 15);
      const int rb  = blockM + waveM * 32 + mt * 16 + ((lane < 16) ? 0 : 8);
      const float bv = bias ? bias[col] : 0.0f;
#pragma unroll
      for (int i = 0; i < 8; ++i) {
        const size_t idx = (size_t)(rb + i) * N + col;
        float v = acc[mt][nt][i] + bv;
        if (mode == 1) v = fmaxf(v, 0.0f);
        if (mode == 2)      Cf[idx] += v;
        else if (mode == 3) Cb[idx]  = (bf16_t)v;
        else                Cf[idx]  = v;
      }
    }
  }
}

// ---------------------------------------------------------------------------
// Weight f32 [K,N] -> bf16 transposed [N,K]  (writes coalesced along K)
// ---------------------------------------------------------------------------
__global__ void transpose_bf16_kernel(const float* __restrict__ s, bf16_t* __restrict__ d,
                                      int K, int N)
{
  long i = (long)blockIdx.x * blockDim.x + threadIdx.x;
  const long total = (long)K * N;
  const long stride = (long)gridDim.x * blockDim.x;
  for (; i < total; i += stride) {
    const int n = (int)(i / K);
    const int k = (int)(i - (long)n * K);
    d[i] = (bf16_t)s[(size_t)k * N + n];
  }
}

// Wo[d]: [16,1024,64] f32 -> transposed-flat bf16 [1024(N=h*64+g), 1024(K=f)]
__global__ void permute_wo_kernel(const float* __restrict__ s, bf16_t* __restrict__ d)
{
  long i = (long)blockIdx.x * blockDim.x + threadIdx.x;
  if (i < 1024L * 1024L) {
    const int f  = (int)(i & 1023);       // K index (contiguous writes)
    const int hg = (int)(i >> 10);        // N index
    const int h  = hg >> 6, g = hg & 63;
    d[i] = (bf16_t)s[((size_t)h * 1024 + f) * 64 + g];
  }
}

// ---------------------------------------------------------------------------
// mask = (sum(v_row) != 0);  qm[row,:] = question[b,:] * mask   (one block/row)
// ---------------------------------------------------------------------------
__global__ __launch_bounds__(256) void build_qm_kernel(
    const float* __restrict__ v, const float* __restrict__ question, float* __restrict__ qm)
{
  __shared__ float red[256];
  const int row = blockIdx.x;
  const int b = row / 100;
  const float* vr = v + (size_t)row * 1024;
  float s = 0.0f;
  for (int c = threadIdx.x; c < 1024; c += 256) s += vr[c];
  red[threadIdx.x] = s;
  __syncthreads();
  for (int off = 128; off > 0; off >>= 1) {
    if (threadIdx.x < off) red[threadIdx.x] += red[threadIdx.x + off];
    __syncthreads();
  }
  const float m = (red[0] != 0.0f) ? 1.0f : 0.0f;
  const float* qrow = question + (size_t)b * 1024;
  float* qo = qm + (size_t)row * 1024;
  for (int c = threadIdx.x; c < 1024; c += 256) qo[c] = qrow[c] * m;
}

// ---------------------------------------------------------------------------
// Attention: block = one (b,r), 512 threads = 16 waves; wave h = head h,
// lane = neighbor n (n<20 active). Softmax via wave32 shfl reductions; then
// nb = att @ xkWo (+bo) accumulated into acc (dir0 writes, dir1 adds).
// ---------------------------------------------------------------------------
__global__ __launch_bounds__(512) void attn_kernel(
    const float* __restrict__ adj, const bf16_t* __restrict__ qv, const bf16_t* __restrict__ kv,
    const float* __restrict__ xkwo, const float* __restrict__ Wbias, const float* __restrict__ bbias,
    const float* __restrict__ bo, float* __restrict__ acc, int dir)
{
  __shared__ float biasv[20];
  __shared__ int   condp[20];
  __shared__ float qs[1024];
  __shared__ float att_s[16 * 24];

  const int t  = threadIdx.x;
  const int br = blockIdx.x;
  const int b  = br / 100;
  const int r  = br - b * 100;

  if (t < 20) {
    const size_t base = (dir == 0)
        ? (((size_t)b * 100 + r) * 100 + t) * 11
        : (((size_t)b * 100 + t) * 100 + r) * 11;
    float s = 0.0f, bv = 0.0f;
#pragma unroll
    for (int l = 0; l < 11; ++l) { const float a = adj[base + l]; s += a; bv += a * Wbias[l]; }
    biasv[t] = bv + bbias[0];
    condp[t] = (s > 0.0f) ? 1 : 0;
  }
  {
    const size_t qb = (size_t)br * 1024;
    qs[t]       = (float)qv[qb + t];
    qs[t + 512] = (float)qv[qb + t + 512];
  }
  __syncthreads();

  const int h = t >> 5;     // head == wave (wave32)
  const int n = t & 31;     // neighbor == lane
  float m;
  if (n < 20) {
    const bf16_t* kr = kv + ((size_t)b * 20 + n) * 1024 + h * 64;
    const float*  qh = qs + h * 64;
    float dot = 0.0f;
#pragma unroll
    for (int j = 0; j < 64; ++j) dot += qh[j] * (float)kr[j];
    m = condp[n] ? dot * 0.125f : -9.0e15f;   // 1/sqrt(64)=0.125
    m += biasv[n];
  } else {
    m = -INFINITY;
  }
  float mx = m;
  for (int off = 16; off > 0; off >>= 1) mx = fmaxf(mx, __shfl_xor(mx, off, 32));
  const float e = (n < 20) ? __expf(m - mx) : 0.0f;
  float sm = e;
  for (int off = 16; off > 0; off >>= 1) sm += __shfl_xor(sm, off, 32);
  if (n < 20) att_s[h * 24 + n] = e / sm;
  __syncthreads();

#pragma unroll
  for (int rep = 0; rep < 2; ++rep) {
    const int f  = t + rep * 512;
    const int h2 = f >> 6;
    float val = bo[f];
    const float* xb = xkwo + (size_t)b * 20 * 1024 + f;
    const float* as = att_s + h2 * 24;
#pragma unroll
    for (int nn = 0; nn < 20; ++nn) val += as[nn] * xb[(size_t)nn * 1024];
    const size_t oidx = (size_t)br * 1024 + f;
    if (dir == 0) acc[oidx] = val;
    else          acc[oidx] += val;
  }
}

// out = v + relu(self + acc)
__global__ void finalize_kernel(const float* __restrict__ v, const float* __restrict__ self,
                                const float* __restrict__ acc, float* __restrict__ out, long n)
{
  long i = (long)blockIdx.x * blockDim.x + threadIdx.x;
  const long stride = (long)gridDim.x * blockDim.x;
  for (; i < n; i += stride) out[i] = v[i] + fmaxf(self[i] + acc[i], 0.0f);
}

// ---------------------------------------------------------------------------
extern "C" void kernel_launch(void* const* d_in, const int* in_sizes, int n_in,
                              void* d_out, int out_size, void* d_ws, size_t ws_size,
                              hipStream_t stream)
{
  (void)in_sizes; (void)n_in; (void)out_size; (void)ws_size;
  const float* visual   = (const float*)d_in[0];
  const float* adj      = (const float*)d_in[1];
  const float* question = (const float*)d_in[2];
  const float* W_v2out  = (const float*)d_in[3];
  const float* b_v2out  = (const float*)d_in[4];
  const float* W_self   = (const float*)d_in[5];
  const float* b_self   = (const float*)d_in[6];
  const float* W_bias   = (const float*)d_in[7];
  const float* b_bias   = (const float*)d_in[8];
  const float* Wq       = (const float*)d_in[9];
  const float* bq       = (const float*)d_in[10];
  const float* Wk       = (const float*)d_in[11];
  const float* bk       = (const float*)d_in[12];
  const float* Wo       = (const float*)d_in[13];
  const float* bo       = (const float*)d_in[14];
  float* out = (float*)d_out;
  char*  ws  = (char*)d_ws;

  const long MR = 25600;       // B*R
  const long MK = 5120;        // B*n (xk rows)
  const long F  = 1024;

  size_t off = 0;
  auto carve = [&](size_t bytes) { size_t p = off; off += (bytes + 255) & ~(size_t)255; return p; };

  float* v_buf  = (float*)(ws + carve(MR * F * 4));
  float* qm     = (float*)(ws + carve(MR * F * 4));
  float* selfb  = (float*)(ws + carve(MR * F * 4));
  float* accb   = (float*)(ws + carve(MR * F * 4));
  bf16_t* qbuf0 = (bf16_t*)(ws + carve(MR * F * 2));
  bf16_t* qbuf1 = (bf16_t*)(ws + carve(MR * F * 2));
  bf16_t* kbuf0 = (bf16_t*)(ws + carve(MK * F * 2));
  bf16_t* kbuf1 = (bf16_t*)(ws + carve(MK * F * 2));
  float* xkwo0  = (float*)(ws + carve(MK * F * 4));
  float* xkwo1  = (float*)(ws + carve(MK * F * 4));
  // All weights stored TRANSPOSED: [N][K] bf16
  bf16_t* wv2o_t  = (bf16_t*)(ws + carve(2048L * 1024 * 2));   // [1024][2048]
  bf16_t* wself_t = (bf16_t*)(ws + carve(2048L * 1024 * 2));   // [1024][2048]
  bf16_t* wq_t    = (bf16_t*)(ws + carve(2L * 1024 * 1024 * 2));
  bf16_t* wk_t    = (bf16_t*)(ws + carve(2L * 1024 * 1024 * 2));
  bf16_t* wo_t    = (bf16_t*)(ws + carve(2L * 1024 * 1024 * 2));

  // 1) weight conversion + transposition
  transpose_bf16_kernel<<<1024, 256, 0, stream>>>(W_v2out, wv2o_t,  2048, 1024);
  transpose_bf16_kernel<<<1024, 256, 0, stream>>>(W_self,  wself_t, 2048, 1024);
  transpose_bf16_kernel<<<1024, 256, 0, stream>>>(Wq,                wq_t,                1024, 1024);
  transpose_bf16_kernel<<<1024, 256, 0, stream>>>(Wq + 1024L * 1024, wq_t + 1024L * 1024, 1024, 1024);
  transpose_bf16_kernel<<<1024, 256, 0, stream>>>(Wk,                wk_t,                1024, 1024);
  transpose_bf16_kernel<<<1024, 256, 0, stream>>>(Wk + 1024L * 1024, wk_t + 1024L * 1024, 1024, 1024);
  permute_wo_kernel<<<4096, 256, 0, stream>>>(Wo,                   wo_t);
  permute_wo_kernel<<<4096, 256, 0, stream>>>(Wo + 16L * 1024 * 64, wo_t + 1024L * 1024);

  // 2) v = relu(visual @ W_v2out + b_v2out)
  gemm_bf16_kernel<<<dim3(MR / BM, F / BN), 256, 0, stream>>>(
      visual, 2048, 0, wv2o_t, 2048, b_v2out, v_buf, nullptr, (int)MR, (int)F, 2048, /*relu*/1);

  // 3) qm = question * mask(v)
  build_qm_kernel<<<MR, 256, 0, stream>>>(v_buf, question, qm);

  // 4) self = v @ W_self[:1024] + b_self ; self += qm @ W_self[1024:]
  //    (transposed weight: K-offset selects the vcq half; row pitch stays 2048)
  gemm_bf16_kernel<<<dim3(MR / BM, F / BN), 256, 0, stream>>>(
      v_buf, 1024, 0, wself_t, 2048, b_self, selfb, nullptr, (int)MR, (int)F, 1024, 0);
  gemm_bf16_kernel<<<dim3(MR / BM, F / BN), 256, 0, stream>>>(
      qm, 1024, 0, wself_t + 1024, 2048, nullptr, selfb, nullptr, (int)MR, (int)F, 1024, /*acc*/2);

  // 5) per-direction projections (q: full rows; k, xkWo: xk row-gather)
  gemm_bf16_kernel<<<dim3(MR / BM, F / BN), 256, 0, stream>>>(
      selfb, 1024, 0, wq_t, 1024, bq, nullptr, qbuf0, (int)MR, (int)F, 1024, /*bf16*/3);
  gemm_bf16_kernel<<<dim3(MR / BM, F / BN), 256, 0, stream>>>(
      selfb, 1024, 0, wq_t + 1024L * 1024, 1024, bq + 1024, nullptr, qbuf1, (int)MR, (int)F, 1024, 3);
  gemm_bf16_kernel<<<dim3(MK / BM, F / BN), 256, 0, stream>>>(
      selfb, 1024, 1, wk_t, 1024, bk, nullptr, kbuf0, (int)MK, (int)F, 1024, 3);
  gemm_bf16_kernel<<<dim3(MK / BM, F / BN), 256, 0, stream>>>(
      selfb, 1024, 1, wk_t + 1024L * 1024, 1024, bk + 1024, nullptr, kbuf1, (int)MK, (int)F, 1024, 3);
  gemm_bf16_kernel<<<dim3(MK / BM, F / BN), 256, 0, stream>>>(
      selfb, 1024, 1, wo_t, 1024, nullptr, xkwo0, nullptr, (int)MK, (int)F, 1024, 0);
  gemm_bf16_kernel<<<dim3(MK / BM, F / BN), 256, 0, stream>>>(
      selfb, 1024, 1, wo_t + 1024L * 1024, 1024, nullptr, xkwo1, nullptr, (int)MK, (int)F, 1024, 0);

  // 6) attention (dir0 writes acc, dir1 accumulates)
  attn_kernel<<<MR, 512, 0, stream>>>(adj, qbuf0, kbuf0, xkwo0, W_bias, b_bias, bo,        accb, 0);
  attn_kernel<<<MR, 512, 0, stream>>>(adj, qbuf1, kbuf1, xkwo1, W_bias, b_bias, bo + 1024, accb, 1);

  // 7) out = v + relu(self + acc)
  finalize_kernel<<<4096, 256, 0, stream>>>(v_buf, selfb, accb, out, MR * F);
}
